// Compatibility_60086592471778
// MI455X (gfx1250) — compile-verified, matrix operand order checked by reading, not run on hardware
//
#include <hip/hip_runtime.h>

// ---------------------------------------------------------------------------
// Problem constants (match the JAX reference)
// ---------------------------------------------------------------------------
#define BATCH 16
#define NQ    2048
#define NK    2048
#define KD    256      // input feature dim
#define KQ    256      // projected dim (K of the score GEMM)
#define CLIP  10.0f
#define SCALE (1.0f / 16.0f)   // 1/sqrt(256)
#define LOG2E 1.44269504088896340736f

typedef __bf16 bf16_t;
typedef __attribute__((ext_vector_type(16))) __bf16 v16bf;
typedef __attribute__((ext_vector_type(8)))  __bf16 v8bf;
typedef __attribute__((ext_vector_type(2)))  __bf16 v2bf;
typedef __attribute__((ext_vector_type(8)))  float  v8f;
typedef int v4i __attribute__((vector_size(4 * sizeof(int))));

#define GLOBAL_AS __attribute__((address_space(1)))
#define LOCAL_AS  __attribute__((address_space(3)))

union Frag16 { v16bf v; v8bf h[2]; };

// ---------------------------------------------------------------------------
// Async global->LDS (CDNA5 ASYNCcnt path), guarded so absence of the builtin
// falls back to a plain 16B vector copy.
// Probe-derived signature: (v4i __device__*, v4i __shared__*, int, int)
// ---------------------------------------------------------------------------
#if __has_builtin(__builtin_amdgcn_global_load_async_to_lds_b128)
#define HAS_ASYNC_LDS 1
#else
#define HAS_ASYNC_LDS 0
#endif

__device__ __forceinline__ void wait_async0() {
#if __has_builtin(__builtin_amdgcn_s_wait_asynccnt)
    __builtin_amdgcn_s_wait_asynccnt(0);
#else
    asm volatile("s_wait_asynccnt 0x0" ::: "memory");
#endif
}

// copy 16 bytes global -> LDS (async when available)
__device__ __forceinline__ void copy16_g2l(bf16_t* ldsDst, const bf16_t* gSrc) {
#if HAS_ASYNC_LDS
    __builtin_amdgcn_global_load_async_to_lds_b128(
        (GLOBAL_AS v4i*)gSrc, (LOCAL_AS v4i*)ldsDst, 0, 0);
#else
    *(v8bf*)ldsDst = *(const v8bf*)gSrc;
#endif
}

__device__ __forceinline__ void copy16_fence() {
#if HAS_ASYNC_LDS
    wait_async0();
#endif
}

// ---------------------------------------------------------------------------
// Fast math helpers (raw CDNA5 transcendental ops when available)
// ---------------------------------------------------------------------------
__device__ __forceinline__ float fast_exp2(float x) {
#if __has_builtin(__builtin_amdgcn_exp2f)
    return __builtin_amdgcn_exp2f(x);      // v_exp_f32
#else
    return exp2f(x);
#endif
}

__device__ __forceinline__ float fast_rcp(float x) {
#if __has_builtin(__builtin_amdgcn_rcpf)
    return __builtin_amdgcn_rcpf(x);       // v_rcp_f32
#else
    return 1.0f / x;
#endif
}

// float -> bf16: let the backend select the native cvt for gfx1250.
__device__ __forceinline__ bf16_t f2bf(float x) {
    return (bf16_t)x;
}

__device__ __forceinline__ v2bf f2bf2(float x, float y) {
    v2bf p;
    p.x = (bf16_t)x;
    p.y = (bf16_t)y;
    return p;
}

// exp(CLIP * tanh(s)) with no branches:
//   tanh(s) = 1 - 2/(e^{2s}+1)  =>  CLIP*tanh(s) = CLIP - 2*CLIP/(e^{2s}+1)
// Saturates correctly: e^{2s}->inf => exp(+CLIP); e^{2s}->0 => exp(-CLIP).
__device__ __forceinline__ float exp_clip_tanh(float s) {
    float p = fast_exp2(s * (2.0f * LOG2E));            // e^{2s}
    float w = CLIP - (2.0f * CLIP) * fast_rcp(p + 1.0f);
    return fast_exp2(w * LOG2E);
}

// Load one wave32 16-bit A/B fragment half-pair for 16x16x32 WMMA.
// `rowPtr` points at element [row][0] of a K-chunk of 32 bf16 values.
// lanes 0..15 take K {0..7, 16..23}; lanes 16..31 take K {8..15, 24..31}.
__device__ __forceinline__ v16bf load_frag32(const bf16_t* rowPtr, int half) {
    Frag16 f;
    f.h[0] = *(const v8bf*)(rowPtr + half * 8);
    f.h[1] = *(const v8bf*)(rowPtr + 16 + half * 8);
    return f.v;
}

// ---------------------------------------------------------------------------
// Kernel 0: transpose + convert Wk/Wq (f32 [d][e]) -> bf16 WT [e][d]
// ---------------------------------------------------------------------------
__global__ __launch_bounds__(256) void wt_kernel(const float* __restrict__ Wk,
                                                 const float* __restrict__ Wq,
                                                 bf16_t* __restrict__ WTk,
                                                 bf16_t* __restrict__ WTq) {
    int idx = blockIdx.x * 256 + threadIdx.x;   // 0 .. 65535
    int e = idx >> 8;
    int d = idx & 255;
    WTk[idx] = f2bf(Wk[d * KQ + e]);
    WTq[idx] = f2bf(Wq[d * KQ + e]);
}

// ---------------------------------------------------------------------------
// Kernel 1: projection  P[b,n,e] = sum_d X[b,n,d] * W[d,e]  (bf16 out)
// Block: 256 threads (8 waves), 64-row x 64-col output tile, K in 32-chunks.
// ---------------------------------------------------------------------------
__global__ __launch_bounds__(256) void proj_kernel(const float*  __restrict__ X,
                                                   const bf16_t* __restrict__ WT,
                                                   bf16_t*       __restrict__ P) {
    __shared__ bf16_t Xs[64][32];
    __shared__ bf16_t Ws[64][32];

    const int b  = blockIdx.z;
    const int r0 = blockIdx.x * 64;
    const int e0 = blockIdx.y * 64;
    const int t    = threadIdx.x;
    const int wave = t >> 5;
    const int lane = t & 31;
    const int half = lane >> 4;
    const int l16  = lane & 15;

    // two C tiles per wave: tile ids (wave) and (wave+8)
    const int mt1 = wave >> 2;        // 0..1
    const int nt  = wave & 3;         // 0..3
    const int mt2 = mt1 + 2;          // 2..3

    v8f c0 = {};
    v8f c1 = {};

    // staging coords: each thread moves 8 elements per chunk
    const int srow = t >> 2;          // 0..63
    const int scol = (t & 3) * 8;     // 0,8,16,24
    const float*  Xbase = X  + ((size_t)(b * NQ + r0 + srow)) * KD;
    const bf16_t* Wbase = WT + (size_t)(e0 + srow) * KD;

    for (int kc = 0; kc < 8; ++kc) {
        __syncthreads();
        // W chunk: already bf16, one 16B copy per thread (async when available)
        copy16_g2l(&Ws[srow][scol], Wbase + kc * 32 + scol);
        // X chunk: f32 -> bf16 conversion while staging
        #pragma unroll
        for (int j = 0; j < 8; j += 2) {
            v2bf p = f2bf2(Xbase[kc * 32 + scol + j], Xbase[kc * 32 + scol + j + 1]);
            *(v2bf*)&Xs[srow][scol + j] = p;
        }
        copy16_fence();
        __syncthreads();

        v16bf bfrag = load_frag32(&Ws[nt * 16 + l16][0], half);
        v16bf a1    = load_frag32(&Xs[mt1 * 16 + l16][0], half);
        v16bf a2    = load_frag32(&Xs[mt2 * 16 + l16][0], half);
        c0 = __builtin_amdgcn_wmma_f32_16x16x32_bf16(false, a1, false, bfrag,
                                                     (short)0, c0, false, false);
        c1 = __builtin_amdgcn_wmma_f32_16x16x32_bf16(false, a2, false, bfrag,
                                                     (short)0, c1, false, false);
    }

    // C layout: VGPR i -> row (i + 8*half_group), lanes 0..15 cols, 16..31 cols
    const int rowAdd = half * 8;
    const int col    = e0 + nt * 16 + l16;
    #pragma unroll
    for (int i = 0; i < 8; ++i) {
        int r = r0 + mt1 * 16 + rowAdd + i;
        P[((size_t)(b * NQ + r)) * KQ + col] = f2bf(c0[i]);
    }
    #pragma unroll
    for (int i = 0; i < 8; ++i) {
        int r = r0 + mt2 * 16 + rowAdd + i;
        P[((size_t)(b * NQ + r)) * KQ + col] = f2bf(c1[i]);
    }
}

// ---------------------------------------------------------------------------
// Kernel 2: fused scores + tanh-clip + mask + softmax
// Block: 512 threads = 16 waves. One block = one batch b, one 16-row M tile,
// full 2048 columns. Each wave owns 128 columns (8 WMMA N-tiles) with the
// exp'd scores register-resident; row sums reduced through LDS ds_add_f32.
// Scores are clipped to [-10,10] before exp, so no row-max pass is needed.
// ---------------------------------------------------------------------------
__global__ __launch_bounds__(512) void attn_kernel(const bf16_t* __restrict__ Q,
                                                   const bf16_t* __restrict__ K,
                                                   const int*    __restrict__ adj,
                                                   float*        __restrict__ out) {
    __shared__ bf16_t qs[16][KQ];
    __shared__ float  sums[16];

    const int b    = blockIdx.y;
    const int m0   = blockIdx.x * 16;
    const int t    = threadIdx.x;
    const int wave = t >> 5;
    const int lane = t & 31;
    const int half = lane >> 4;
    const int l16  = lane & 15;

    // stage the 16x256 q tile (one 16B transfer per thread)
    {
        int row = t >> 5;              // 0..15
        int col = (t & 31) * 8;        // 0..248
        copy16_g2l(&qs[row][col],
                   Q + ((size_t)(b * NQ + m0 + row)) * KQ + col);
    }
    if (t < 16) sums[t] = 0.0f;
    copy16_fence();
    __syncthreads();

    // preload the 8 A fragments (same q tile for every column tile)
    Frag16 a[8];
    const bf16_t* qrow = &qs[l16][0];
    #pragma unroll
    for (int kc = 0; kc < 8; ++kc) {
        a[kc].h[0] = *(const v8bf*)(qrow + kc * 32 + half * 8);
        a[kc].h[1] = *(const v8bf*)(qrow + kc * 32 + 16 + half * 8);
    }

    const int nbase = wave * 128;
    const bf16_t* Kb = K + (size_t)b * NK * KQ;

    v8f c[8];
    #pragma unroll
    for (int tt = 0; tt < 8; ++tt) {
        const int n0 = nbase + tt * 16;
        const bf16_t* krow = Kb + (size_t)(n0 + l16) * KQ;
        // hide the strided-row gather latency of the NEXT tile
        if (tt < 7)
            __builtin_prefetch(krow + 16 * KQ, 0, 1);   // global_prefetch_b8
        v8f acc = {};
        #pragma unroll
        for (int kc = 0; kc < 8; ++kc) {
            Frag16 bf_;
            bf_.h[0] = *(const v8bf*)(krow + kc * 32 + half * 8);
            bf_.h[1] = *(const v8bf*)(krow + kc * 32 + 16 + half * 8);
            acc = __builtin_amdgcn_wmma_f32_16x16x32_bf16(false, a[kc].v, false,
                                                          bf_.v, (short)0, acc,
                                                          false, false);
        }
        c[tt] = acc;
    }

    // tanh-clip + exp fused (3 trans ops), mask, accumulate per-lane row sums.
    const int mgrp = half * 8;                       // this lane's row group
    const int coln = l16;
    const int* adjbase = adj + ((size_t)(b) * NQ + m0) * (size_t)NK;
    float rsum[8];
    #pragma unroll
    for (int i = 0; i < 8; ++i) rsum[i] = 0.0f;

    #pragma unroll
    for (int tt = 0; tt < 8; ++tt) {
        const int n = nbase + tt * 16 + coln;
        #pragma unroll
        for (int i = 0; i < 8; ++i) {
            const int ml = mgrp + i;
            float s = c[tt][i] * SCALE;
            int   am = adjbase[(size_t)ml * NK + n];
            float e = am ? exp_clip_tanh(s) : 0.0f;
            c[tt][i] = e;
            rsum[i] += e;
        }
    }
    #pragma unroll
    for (int i = 0; i < 8; ++i) atomicAdd(&sums[mgrp + i], rsum[i]);
    __syncthreads();

    float rinv[8];
    #pragma unroll
    for (int i = 0; i < 8; ++i) rinv[i] = fast_rcp(sums[mgrp + i]);

    float* orow = out + ((size_t)(b) * NQ + m0) * (size_t)NK;
    #pragma unroll
    for (int tt = 0; tt < 8; ++tt) {
        const int n = nbase + tt * 16 + coln;
        #pragma unroll
        for (int i = 0; i < 8; ++i)
            orow[(size_t)(mgrp + i) * NK + n] = c[tt][i] * rinv[i];
    }
}

// ---------------------------------------------------------------------------
// Host launcher
// ---------------------------------------------------------------------------
extern "C" void kernel_launch(void* const* d_in, const int* in_sizes, int n_in,
                              void* d_out, int out_size, void* d_ws, size_t ws_size,
                              hipStream_t stream) {
    const float* k_in = (const float*)d_in[0];   // [B, NK, KD]
    const float* q_in = (const float*)d_in[1];   // [B, NQ, QD]
    const int*   adj  = (const int*)d_in[2];     // [B, NQ, NK]
    const float* Wk   = (const float*)d_in[3];   // [KD, KQ]
    const float* Wq   = (const float*)d_in[4];   // [QD, KQ]
    float* out = (float*)d_out;                  // [B, NQ, NK]

    const size_t projBytes = (size_t)BATCH * NQ * KQ * sizeof(bf16_t);  // 16 MB
    char* ws = (char*)d_ws;
    bf16_t* qproj = (bf16_t*)ws;
    bf16_t* kproj = (bf16_t*)(ws + projBytes);
    bf16_t* WTk   = (bf16_t*)(ws + 2 * projBytes);
    bf16_t* WTq   = WTk + (size_t)KD * KQ;

    // 0) transpose+convert weights to bf16 [e][d]
    wt_kernel<<<256, 256, 0, stream>>>(Wk, Wq, WTk, WTq);

    // 1) projections (bf16 WMMA)
    dim3 pgrid(NQ / 64, KQ / 64, BATCH);   // (32, 4, 16)
    proj_kernel<<<pgrid, 256, 0, stream>>>(k_in, WTk, kproj);
    proj_kernel<<<pgrid, 256, 0, stream>>>(q_in, WTq, qproj);

    // 2) fused scores + masked softmax
    dim3 agrid(NQ / 16, BATCH);            // (128, 16)
    attn_kernel<<<agrid, 512, 0, stream>>>(qproj, kproj, adj, out);
}
